// MultiLayerRNN_188978561430
// MI455X (gfx1250) — compile-verified, matrix operand order checked by reading, not run on hardware
//
#include <hip/hip_runtime.h>
#include <math.h>

// Problem constants (match reference)
#define B_   64
#define T_   1024
#define IN_  1024
#define H_   1024
#define NBLK 32     // persistent kernel blocks (32*8 waves = 256 tiles)
#define NTHR 256    // 8 wave32 per block

typedef __bf16 bf16;
typedef __bf16 v8bf  __attribute__((ext_vector_type(8)));
typedef __bf16 v16bf __attribute__((ext_vector_type(16)));
typedef float  v4f   __attribute__((ext_vector_type(4)));
typedef float  v8f   __attribute__((ext_vector_type(8)));

#define NW1 (1024L * 1024L)  // elements per weight matrix

#define WMMA_BF16(a, b, c)                                                  \
  __builtin_amdgcn_wmma_f32_16x16x32_bf16(false, (a), false, (b), (short)0, \
                                          (c), false, false)

// ---------------------------------------------------------------------------
// Fast f32 tanh from hardware transcendentals: v_exp_f32 + v_rcp_f32.
// |err| ~1e-6, negligible vs the bf16 state rounding already accepted.
// ---------------------------------------------------------------------------
__device__ __forceinline__ float fast_tanh(float x) {
  float cx = fminf(fmaxf(x, -9.0f), 9.0f);
  float e = __builtin_amdgcn_exp2f(cx * 2.885390081777927f);  // e^(2x)
  return (e - 1.0f) * __builtin_amdgcn_rcpf(e + 1.0f);
}

// ---------------------------------------------------------------------------
// Build a v16bf WMMA operand for lane (lo,hi) from a bf16 row pointer.
// 16-bit operand layout (ISA 7.12.2): VGPRs 0..3 hold K = 8*hi+0..7,
// VGPRs 4..7 hold K = 16+8*hi+0..7  -> two aligned 16B loads.
// ---------------------------------------------------------------------------
__device__ __forceinline__ v16bf ld_op_bf16(const bf16* row, int hi) {
  v8bf c0 = *(const v8bf*)(row + 8 * hi);
  v8bf c1 = *(const v8bf*)(row + 16 + 8 * hi);
  return __builtin_shufflevector(c0, c1, 0, 1, 2, 3, 4, 5, 6, 7, 8, 9, 10, 11,
                                 12, 13, 14, 15);
}

// Same operand, but source row is f32 (converted in-register).
__device__ __forceinline__ v16bf ld_op_f32(const float* row, int hi) {
  const float* p0 = row + 8 * hi;
  const float* p1 = row + 16 + 8 * hi;
  v4f a0 = *(const v4f*)p0;
  v4f a1 = *(const v4f*)(p0 + 4);
  v4f b0 = *(const v4f*)p1;
  v4f b1 = *(const v4f*)(p1 + 4);
  v16bf r;
#pragma unroll
  for (int j = 0; j < 4; ++j) {
    r[j]      = (bf16)a0[j];
    r[4 + j]  = (bf16)a1[j];
    r[8 + j]  = (bf16)b0[j];
    r[12 + j] = (bf16)b1[j];
  }
  return r;
}

// ---------------------------------------------------------------------------
// C += A(16xK) * B(Kx16) with bf16 A/B (B[k][n] = W[n][k], W row-major).
// ---------------------------------------------------------------------------
__device__ __forceinline__ void gemm_bf16(const bf16* __restrict__ A0, long lda,
                                          const bf16* __restrict__ B0, long ldb,
                                          int K, v8f& c) {
  const int lane = threadIdx.x & 31;
  const int lo = lane & 15;
  const int hi = lane >> 4;
  const bf16* pa = A0 + (long)lo * lda;
  const bf16* pb = B0 + (long)lo * ldb;
#pragma unroll 4
  for (int k = 0; k < K; k += 32) {
    v16bf a = ld_op_bf16(pa, hi);
    v16bf b = ld_op_bf16(pb, hi);
    c = WMMA_BF16(a, b, c);
    pa += 32;
    pb += 32;
  }
}

// Store 16x16 f32 C/D tile: VGPR r -> row (r + 8*hi), col lo
__device__ __forceinline__ void store_tile_f32(float* __restrict__ out, long ld,
                                               long m0, long n0, const v8f& c) {
  const int lane = threadIdx.x & 31;
  const int lo = lane & 15;
  const int hi = lane >> 4;
#pragma unroll
  for (int r = 0; r < 8; ++r)
    out[(m0 + r + 8 * hi) * ld + n0 + lo] = c[r];
}

__device__ __forceinline__ void store_tile_bf16(bf16* __restrict__ out, long ld,
                                                long m0, long n0,
                                                const v8f& c) {
  const int lane = threadIdx.x & 31;
  const int lo = lane & 15;
  const int hi = lane >> 4;
#pragma unroll
  for (int r = 0; r < 8; ++r)
    out[(m0 + r + 8 * hi) * ld + n0 + lo] = (bf16)c[r];
}

// ---------------------------------------------------------------------------
// Software grid barrier (NBLK blocks resident). bar[0]=counter, bar[1]=gen.
// ---------------------------------------------------------------------------
__device__ __forceinline__ void grid_barrier(unsigned* bar) {
  __syncthreads();
  if (threadIdx.x == 0) {
    unsigned gen =
        __hip_atomic_load(&bar[1], __ATOMIC_RELAXED, __HIP_MEMORY_SCOPE_AGENT);
    unsigned prev = __hip_atomic_fetch_add(&bar[0], 1u, __ATOMIC_ACQ_REL,
                                           __HIP_MEMORY_SCOPE_AGENT);
    if (prev == NBLK - 1) {
      __hip_atomic_store(&bar[0], 0u, __ATOMIC_RELAXED,
                         __HIP_MEMORY_SCOPE_AGENT);
      __hip_atomic_store(&bar[1], gen + 1u, __ATOMIC_RELEASE,
                         __HIP_MEMORY_SCOPE_AGENT);
    } else {
      while (__hip_atomic_load(&bar[1], __ATOMIC_ACQUIRE,
                               __HIP_MEMORY_SCOPE_AGENT) == gen) {
        __builtin_amdgcn_s_sleep(8);
      }
    }
  }
  __syncthreads();
}

// ---------------------------------------------------------------------------
// Kernel 0: f32 -> bf16 weight conversion (8 MB, L2-resident afterwards)
// and grid-barrier re-init (ws is poisoned once by the harness).
// Wb layout: [Wih0 | Wih1 | Whh0 | Whh1], each 1024x1024 row-major.
// ---------------------------------------------------------------------------
__global__ void rnn_convert_weights(const float* __restrict__ Wih,
                                    const float* __restrict__ Whh,
                                    bf16* __restrict__ Wb,
                                    unsigned* __restrict__ bar) {
  if (blockIdx.x == 0 && threadIdx.x == 0) {
    bar[0] = 0u;
    bar[1] = 0u;
  }
  const long tid = (long)blockIdx.x * blockDim.x + threadIdx.x;
  const long nthr = (long)gridDim.x * blockDim.x;
  for (long i = tid; i < 2 * NW1; i += nthr) Wb[i] = (bf16)Wih[i];
  for (long i = tid; i < 2 * NW1; i += nthr) Wb[2 * NW1 + i] = (bf16)Whh[i];
}

// ---------------------------------------------------------------------------
// Kernel 1: fully parallel precompute  U0[t,b,:] = x[b,t,:] @ Wih0^T + bih0
// (f32 result staged in the ys region of d_out; x converted bf16 in-register)
// ---------------------------------------------------------------------------
__global__ void rnn_precompute_u0(const float* __restrict__ x,
                                  const bf16* __restrict__ Wih0b,
                                  const float* __restrict__ bih0,
                                  float* __restrict__ u0) {
  const int lane = threadIdx.x & 31;
  const int lo = lane & 15;
  const int hi = lane >> 4;
  const int wave = (int)((blockIdx.x * blockDim.x + threadIdx.x) >> 5);
  const int nwaves = (int)((gridDim.x * blockDim.x) >> 5);
  const int ntiles = T_ * (B_ / 16) * (H_ / 16);  // 262144 tiles
  for (int tile = wave; tile < ntiles; tile += nwaves) {
    const int n = tile & 63;         // H tile
    const int bm = (tile >> 6) & 3;  // batch tile
    const int t = tile >> 8;         // timestep
    v8f c = {};
    const float* pa =
        x + (long)(bm * 16 + lo) * T_ * IN_ + (long)t * IN_;  // A row (f32)
    const bf16* pb = Wih0b + (long)(n * 16 + lo) * IN_;       // W row (bf16)
#pragma unroll 4
    for (int k = 0; k < IN_; k += 32) {
      v16bf a = ld_op_f32(pa, hi);
      v16bf b = ld_op_bf16(pb, hi);
      c = WMMA_BF16(a, b, c);
      pa += 32;
      pb += 32;
    }
    const float bias = bih0[n * 16 + lo];
#pragma unroll
    for (int r = 0; r < 8; ++r) c[r] += bias;
    store_tile_f32(u0 + (long)t * B_ * H_, H_, bm * 16, n * 16, c);
  }
}

// ---------------------------------------------------------------------------
// Kernel 2: persistent, software-pipelined recurrence.
// Phase p (p = 0..T): layer0(t=p) and layer1(t=p-1) run fused in ONE K-loop
// (shared h0[p-1] A-operand, two accumulators) -> one grid barrier per phase.
// Tile map m0=(g&3), n0=(g>>2): the 4 m-tiles of an n-tile are adjacent waves
// of one workgroup, so weight cachelines are reused out of the WGP cache.
// ---------------------------------------------------------------------------
__global__ void __launch_bounds__(NTHR)
rnn_recurrent(const float* __restrict__ h0in,  // (B, L, H) initial state (f32)
              const bf16* __restrict__ Wih1b, const float* __restrict__ bih1,
              const bf16* __restrict__ Whh0b, const float* __restrict__ bhh0,
              const bf16* __restrict__ Whh1b, const float* __restrict__ bhh1,
              float* __restrict__ out,   // (T,B,H): holds U0, becomes ys
              float* __restrict__ hfin,  // (B,L,H) final hidden (f32)
              bf16* __restrict__ state,  // 4 * B*H bf16 ping/pong
              unsigned* __restrict__ bar) {
  bf16* h0buf[2] = {state, state + (long)B_ * H_};
  bf16* h1buf[2] = {state + 2L * B_ * H_, state + 3L * B_ * H_};

  const int tid = (int)(blockIdx.x * blockDim.x + threadIdx.x);
  const int nthr = NBLK * NTHR;

  // Initial states h0[-1], h1[-1] live in the parity-1 buffers.
  for (int i = tid; i < B_ * H_; i += nthr) {
    const int b = i >> 10;  // / H_
    const int h = i & (H_ - 1);
    h0buf[1][i] = (bf16)h0in[(long)b * 2 * H_ + h];
    h1buf[1][i] = (bf16)h0in[(long)b * 2 * H_ + H_ + h];
  }
  grid_barrier(bar);

  const int lane = threadIdx.x & 31;
  const int lo = lane & 15;
  const int hi = lane >> 4;
  const int g = tid >> 5;               // wave id 0..255 == tile id
  const long m0 = (long)(g & 3) * 16;   // batch tile (adjacent waves share n)
  const long n0 = (long)(g >> 2) * 16;  // hidden tile

  const float bias0 = bhh0[n0 + lo];
  const float bias1 = bih1[n0 + lo] + bhh1[n0 + lo];

  // Per-lane row bases (constant across phases except the state buffer swap)
  const long arow = (m0 + lo) * H_;  // A-operand row offset within state
  const long brow = (n0 + lo) * H_;  // B-operand row offset within weights
  const bf16* pb0base = Whh0b + brow;
  const bf16* pb1base = Wih1b + brow;
  const bf16* pb2base = Whh1b + brow;

  for (int p = 0; p <= T_; ++p) {
    const bf16* h0rd = h0buf[(p + 1) & 1];  // h0[p-1]
    bf16* h0wr = h0buf[p & 1];              // h0[p]
    const bf16* h1rd = h1buf[p & 1];        // h1[p-2]
    bf16* h1wr = h1buf[(p + 1) & 1];        // h1[p-1]

    // Prefetch next phase's U0 tile (one 64B line per row).
    if (p + 1 < T_) {
      const float* un = out + (long)(p + 1) * B_ * H_;
      __builtin_prefetch(&un[(m0 + lo) * H_ + n0], 0, 1);
    }

    if (p == 0) {
      // ---- prologue: layer0(t=0) only
      const float* u = out;
      v8f c;
#pragma unroll
      for (int r = 0; r < 8; ++r) c[r] = u[(m0 + r + 8 * hi) * H_ + n0 + lo];
      gemm_bf16(h0rd + m0 * H_, H_, Whh0b + n0 * H_, H_, H_, c);
#pragma unroll
      for (int r = 0; r < 8; ++r) c[r] = fast_tanh(c[r] + bias0);
      store_tile_bf16(h0wr, H_, m0, n0, c);
    } else if (p == T_) {
      // ---- epilogue: layer1(t=T-1) only
      v8f d = {};
      gemm_bf16(h0rd + m0 * H_, H_, Wih1b + n0 * H_, H_, H_, d);
      gemm_bf16(h1rd + m0 * H_, H_, Whh1b + n0 * H_, H_, H_, d);
#pragma unroll
      for (int r = 0; r < 8; ++r) d[r] = fast_tanh(d[r] + bias1);
      store_tile_f32(out + (long)(T_ - 1) * B_ * H_, H_, m0, n0, d);
      store_tile_bf16(h1wr, H_, m0, n0, d);
#pragma unroll
      for (int r = 0; r < 8; ++r)  // final layer-1 hidden state (f32)
        hfin[(m0 + r + 8 * hi) * 2 * H_ + H_ + n0 + lo] = d[r];
    } else {
      // ---- fused steady state: layer0(t=p) + layer1(t=p-1), one K-loop
      const float* u = out + (long)p * B_ * H_;
      v8f c;  // layer0 accumulator, seeded with U0[p]
#pragma unroll
      for (int r = 0; r < 8; ++r) c[r] = u[(m0 + r + 8 * hi) * H_ + n0 + lo];
      v8f d = {};  // layer1 accumulator

      const bf16* pa0 = h0rd + arow;  // shared A: h0[p-1]
      const bf16* pa1 = h1rd + arow;  // A: h1[p-2]
      const bf16* pb0 = pb0base;      // Whh0
      const bf16* pb1 = pb1base;      // Wih1
      const bf16* pb2 = pb2base;      // Whh1
#pragma unroll 2
      for (int k = 0; k < H_; k += 32) {
        v16bf a0 = ld_op_bf16(pa0, hi);
        v16bf a1 = ld_op_bf16(pa1, hi);
        v16bf b0 = ld_op_bf16(pb0, hi);
        v16bf b1 = ld_op_bf16(pb1, hi);
        v16bf b2 = ld_op_bf16(pb2, hi);
        c = WMMA_BF16(a0, b0, c);  // h0[p-1] @ Whh0^T
        d = WMMA_BF16(a0, b1, d);  // h0[p-1] @ Wih1^T
        d = WMMA_BF16(a1, b2, d);  // h1[p-2] @ Whh1^T
        pa0 += 32; pa1 += 32; pb0 += 32; pb1 += 32; pb2 += 32;
      }

      // layer0 epilogue
#pragma unroll
      for (int r = 0; r < 8; ++r) c[r] = fast_tanh(c[r] + bias0);
      store_tile_bf16(h0wr, H_, m0, n0, c);
      if (p == T_ - 1) {  // final layer-0 hidden state (f32)
#pragma unroll
        for (int r = 0; r < 8; ++r)
          hfin[(m0 + r + 8 * hi) * 2 * H_ + n0 + lo] = c[r];
      }
      // layer1 epilogue
#pragma unroll
      for (int r = 0; r < 8; ++r) d[r] = fast_tanh(d[r] + bias1);
      store_tile_f32(out + (long)(p - 1) * B_ * H_, H_, m0, n0, d);
      store_tile_bf16(h1wr, H_, m0, n0, d);
    }
    grid_barrier(bar);
  }
}

// ---------------------------------------------------------------------------
extern "C" void kernel_launch(void* const* d_in, const int* in_sizes, int n_in,
                              void* d_out, int out_size, void* d_ws,
                              size_t ws_size, hipStream_t stream) {
  (void)in_sizes; (void)n_in; (void)out_size; (void)ws_size;
  const float* x   = (const float*)d_in[0];  // (B,T,IN)
  const float* h0  = (const float*)d_in[1];  // (B,L,H)
  const float* Wih = (const float*)d_in[2];  // (L,H,IN)
  const float* bih = (const float*)d_in[3];  // (L,H)
  const float* Whh = (const float*)d_in[4];  // (L,H,H)
  const float* bhh = (const float*)d_in[5];  // (L,H)
  float* out = (float*)d_out;                // T*B*H (ys) + B*L*H (h_final)

  // ws layout: [state: 4*B*H bf16 = 512KB][Wb: 4*1M bf16 = 8MB][barrier]
  bf16* state = (bf16*)d_ws;
  bf16* Wb = state + 4L * B_ * H_;
  unsigned* bar = (unsigned*)(Wb + 4 * NW1);

  // Phase 0: weight conversion to bf16 (+ barrier re-init)
  rnn_convert_weights<<<dim3(512), dim3(NTHR), 0, stream>>>(Wih, Whh, Wb, bar);
  // Phase 1: wide parallel GEMM for layer-0 input projection
  rnn_precompute_u0<<<dim3(2048), dim3(NTHR), 0, stream>>>(x, Wb, bih, out);
  // Phase 2: persistent software-pipelined recurrence (1 barrier per step)
  rnn_recurrent<<<dim3(NBLK), dim3(NTHR), 0, stream>>>(
      h0, Wb + NW1, bih + H_, Wb + 2 * NW1, bhh, Wb + 3 * NW1, bhh + H_, out,
      out + (long)T_ * B_ * H_, state, bar);
}
